// GPTAttention_13864154431645
// MI455X (gfx1250) — matrix operand order repaired and checked by reading, not compiled, not run on hardware
//
#include <hip/hip_runtime.h>

#define D_MODEL 768
#define N_HEADS 12
#define D_HEAD  64
#define SEQ     2048
#define BATCH   2
#define NTOK    (BATCH*SEQ)          // 4096
#define QKV_N   (3*D_MODEL)          // 2304

typedef __attribute__((ext_vector_type(16))) _Float16 v16h;
typedef __attribute__((ext_vector_type(8)))  _Float16 v8h;
typedef __attribute__((ext_vector_type(8)))  float    v8f;

__device__ __forceinline__ v16h combine16(v8h lo, v8h hi) {
  return __builtin_shufflevector(lo, hi, 0,1,2,3,4,5,6,7,8,9,10,11,12,13,14,15);
}

// ---- CDNA5 async memory->LDS copy (ASYNCcnt path, no VGPR transit) ----
__device__ __forceinline__ void async_copy_b128(void* lds, const void* gptr) {
  unsigned int laddr = (unsigned int)(size_t)lds;   // generic LDS ptr: low 32 bits = LDS address
  asm volatile("global_load_async_to_lds_b128 %0, %1, off"
               :: "v"(laddr), "v"(gptr)
               : "memory");
}
__device__ __forceinline__ void wait_async() {
  asm volatile("s_wait_asynccnt 0" ::: "memory");
}

// A-matrix fragment (16xK=32 f16). lane<16: row=lane, K=0..7 & 16..23.
// lane>=16: row=lane-16, K=8..15 & 24..31.  `stride` in halves.
__device__ __forceinline__ v16h load_frag_a(const _Float16* base, int stride) {
  int lane = threadIdx.x & 31;
  int row  = lane & 15;
  int koff = (lane >> 4) << 3;            // 0 or 8
  const _Float16* p = base + row * stride + koff;
  v8h lo = *(const v8h*)(p);
  v8h hi = *(const v8h*)(p + 16);
  return combine16(lo, hi);
}

// B-matrix fragment (K=32 x 16 f16), stored "column-major" in memory:
// each output column's 32 K-values contiguous. lane: col=lane&15,
// K=(lane<16 ? 0..7 & 16..23 : 8..15 & 24..31) packed 2 per VGPR.
__device__ __forceinline__ v16h load_frag_b(const _Float16* base, int stride) {
  int lane  = threadIdx.x & 31;
  int col   = lane & 15;
  int kbase = (lane >> 4) << 3;           // 0 or 8
  const _Float16* p = base + col * stride + kbase;
  v8h lo = *(const v8h*)(p);
  v8h hi = *(const v8h*)(p + 16);
  return combine16(lo, hi);
}

#define WMMA(a, b, c) __builtin_amdgcn_wmma_f32_16x16x32_f16(false, (a), false, (b), (short)0, (c), false, false)

// ---------------- stage 1: conversions ----------------
__global__ void cvt_f32_f16(const float* __restrict__ src, _Float16* __restrict__ dst, int n) {
  int i = blockIdx.x * blockDim.x + threadIdx.x;
  if (i < n) dst[i] = (_Float16)src[i];
}

// WT[n*K + k] = W[k*N + n]  (f32 -> f16 transpose, N-major output)
__global__ void cvt_transpose(const float* __restrict__ W, _Float16* __restrict__ WT, int K, int N) {
  int i = blockIdx.x * blockDim.x + threadIdx.x;
  if (i < K * N) {
    int k = i / N, n = i % N;           // coalesced reads along n
    WT[(size_t)n * K + k] = (_Float16)W[(size_t)k * N + n];
  }
}

// ---------------- stage 2: QKV GEMM ----------------
// C[4096,2304] = xh @ Wqkv + bqkv ; scatter into Qh/Kh/Vh [B,H,S,64] f16
__global__ __launch_bounds__(256) void qkv_gemm(
    const _Float16* __restrict__ xh, const _Float16* __restrict__ WT,
    const float* __restrict__ bias,
    _Float16* __restrict__ Qh, _Float16* __restrict__ Kh, _Float16* __restrict__ Vh) {
  __shared__ _Float16 sA[2][128 * 40];
  __shared__ _Float16 sB[2][128 * 40];
  const int m0 = blockIdx.x * 128;
  const int n0 = blockIdx.y * 128;
  const int tid = threadIdx.x;
  const int wave = tid >> 5, lane = tid & 31;
  const int wm = (wave & 3) * 32;        // wave row offset in tile
  const int wn = (wave >> 2) * 64;       // wave col offset in tile

  v8f acc[2][4] = {};

  const int srow = tid >> 1, sseg = (tid & 1) * 16;
  auto stage = [&](int buf, int k0) {
    const _Float16* ga = xh + (size_t)(m0 + srow) * D_MODEL + k0 + sseg;
    _Float16* la = &sA[buf][srow * 40 + sseg];
    async_copy_b128(la,     ga);
    async_copy_b128(la + 8, ga + 8);
    const _Float16* gb = WT + (size_t)(n0 + srow) * D_MODEL + k0 + sseg;
    _Float16* lb = &sB[buf][srow * 40 + sseg];
    async_copy_b128(lb,     gb);
    async_copy_b128(lb + 8, gb + 8);
  };

  stage(0, 0);
  int cur = 0;
  for (int k0 = 0; k0 < D_MODEL; k0 += 32) {
    wait_async();           // tile [cur] resident in LDS
    __syncthreads();        // all waves staged; prior reads of [cur^1] done
    if (k0 + 32 < D_MODEL) stage(cur ^ 1, k0 + 32);   // prefetch overlaps compute
    v16h a0 = load_frag_a(&sA[cur][(wm + 0) * 40], 40);
    v16h a1 = load_frag_a(&sA[cur][(wm + 16) * 40], 40);
    #pragma unroll
    for (int j = 0; j < 4; ++j) {
      v16h b = load_frag_b(&sB[cur][(wn + j * 16) * 40], 40);
      acc[0][j] = WMMA(a0, b, acc[0][j]);
      acc[1][j] = WMMA(a1, b, acc[1][j]);
    }
    cur ^= 1;
  }

  const int col = lane & 15;
  const int rhalf = (lane >> 4) * 8;
  #pragma unroll
  for (int i = 0; i < 2; ++i)
    #pragma unroll
    for (int j = 0; j < 4; ++j)
      #pragma unroll
      for (int r = 0; r < 8; ++r) {
        int m = m0 + wm + i * 16 + rhalf + r;      // token
        int n = n0 + wn + j * 16 + col;            // qkv column
        float v = acc[i][j][r] + bias[n];
        int seg = n / D_MODEL;                      // 0=q 1=k 2=v
        int nn = n - seg * D_MODEL;
        int h = nn >> 6, d = nn & 63;
        int b = m >> 11, s = m & 2047;
        size_t idx = ((((size_t)b * N_HEADS + h) * SEQ) + s) * D_HEAD + d;
        if (seg == 0)      Qh[idx] = (_Float16)(v * 0.125f);   // fold 1/sqrt(64)
        else if (seg == 1) Kh[idx] = (_Float16)v;
        else               Vh[idx] = (_Float16)v;
      }
}

// ---------------- stage 3: causal flash attention ----------------
__global__ __launch_bounds__(256) void attn_fwd(
    const _Float16* __restrict__ Qh, const _Float16* __restrict__ Kh,
    const _Float16* __restrict__ Vh, _Float16* __restrict__ attnh) {
  __shared__ _Float16 sK[128 * 72];      // [key][d]
  __shared__ _Float16 sV[64 * 136];      // [d][key] (transposed)
  __shared__ _Float16 sP[8 * 16 * 40];   // per-wave P buffer [16 q][32 k]

  const int qTile = blockIdx.x;          // 16 tiles of 128 queries
  const int bh = blockIdx.y;             // b*12+h
  const size_t base = (size_t)bh * SEQ * D_HEAD;
  const int tid = threadIdx.x, wave = tid >> 5, lane = tid & 31;
  const int qRow = qTile * 128 + wave * 16;
  const int col = lane & 15;
  const int rhalf = (lane >> 4) * 8;
  _Float16* pbuf = sP + wave * 16 * 40;

  // Q fragments held in registers for the whole kernel (scale pre-folded)
  v16h qf0 = load_frag_a(Qh + base + (size_t)qRow * D_HEAD + 0, D_HEAD);
  v16h qf1 = load_frag_a(Qh + base + (size_t)qRow * D_HEAD + 32, D_HEAD);

  v8f o[4] = {};
  float rm[8], rl[8];
  #pragma unroll
  for (int r = 0; r < 8; ++r) { rm[r] = -__builtin_inff(); rl[r] = 0.0f; }

  const int skey = tid >> 1, sseg = (tid & 1) * 32;
  for (int kt = 0; kt <= qTile; ++kt) {
    __syncthreads();                      // prior tile's reads complete
    { // K tile: async memory->LDS (layout-preserving, padded stride)
      const _Float16* gk = Kh + base + (size_t)(kt * 128 + skey) * D_HEAD + sseg;
      _Float16* lk = sK + skey * 72 + sseg;
      async_copy_b128(lk,      gk);
      async_copy_b128(lk + 8,  gk + 8);
      async_copy_b128(lk + 16, gk + 16);
      async_copy_b128(lk + 24, gk + 24);
      // V tile transposed [d][key] via VGPR scatter
      const float4* sv = (const float4*)(Vh + base + (size_t)(kt * 128 + skey) * D_HEAD + sseg);
      _Float16 tmp[32];
      *(float4*)&tmp[0]  = sv[0]; *(float4*)&tmp[8]  = sv[1];
      *(float4*)&tmp[16] = sv[2]; *(float4*)&tmp[24] = sv[3];
      #pragma unroll
      for (int d2 = 0; d2 < 32; ++d2)
        sV[(sseg + d2) * 136 + skey] = tmp[d2];
    }
    wait_async();
    __syncthreads();

    const bool diag = (kt == qTile);
    #pragma unroll
    for (int kb = 0; kb < 4; ++kb) {
      const int k0 = kb * 32;
      // scores: two 16-key column blocks, contraction over d=64 (2 chunks)
      v8f s0 = {}, s1 = {};
      s0 = WMMA(qf0, load_frag_b(sK + (k0 + 0) * 72 + 0, 72), s0);
      s0 = WMMA(qf1, load_frag_b(sK + (k0 + 0) * 72 + 32, 72), s0);
      s1 = WMMA(qf0, load_frag_b(sK + (k0 + 16) * 72 + 0, 72), s1);
      s1 = WMMA(qf1, load_frag_b(sK + (k0 + 16) * 72 + 32, 72), s1);

      const int kg0 = kt * 128 + k0 + col;
      #pragma unroll
      for (int r = 0; r < 8; ++r) {
        const int qg = qRow + rhalf + r;
        float v0 = s0[r], v1 = s1[r];
        if (diag) {
          if (kg0 > qg)      v0 = -__builtin_inff();
          if (kg0 + 16 > qg) v1 = -__builtin_inff();
        }
        float mx = fmaxf(v0, v1);
        mx = fmaxf(mx, __shfl_xor(mx, 1, 32));
        mx = fmaxf(mx, __shfl_xor(mx, 2, 32));
        mx = fmaxf(mx, __shfl_xor(mx, 4, 32));
        mx = fmaxf(mx, __shfl_xor(mx, 8, 32));
        const float mnew = fmaxf(rm[r], mx);
        const float alpha = __expf(rm[r] - mnew);
        const float p0 = __expf(v0 - mnew);
        const float p1 = __expf(v1 - mnew);
        float sum = p0 + p1;
        sum += __shfl_xor(sum, 1, 32);
        sum += __shfl_xor(sum, 2, 32);
        sum += __shfl_xor(sum, 4, 32);
        sum += __shfl_xor(sum, 8, 32);
        rl[r] = rl[r] * alpha + sum;
        rm[r] = mnew;
        o[0][r] *= alpha; o[1][r] *= alpha; o[2][r] *= alpha; o[3][r] *= alpha;
        const int prow = rhalf + r;
        pbuf[prow * 40 + col]      = (_Float16)p0;
        pbuf[prow * 40 + 16 + col] = (_Float16)p1;
      }
      // P@V : A=P [16x32], B=V^T chunks (contiguous keys per d column)
      v16h pf = load_frag_a(pbuf, 40);   // DS ops are in-order per wave
      #pragma unroll
      for (int dc = 0; dc < 4; ++dc) {
        v16h vf = load_frag_b(sV + (dc * 16) * 136 + k0, 136);
        o[dc] = WMMA(pf, vf, o[dc]);
      }
    }
  }

  // epilogue: normalize and write [token, h*64+d] f16
  const int b = bh / N_HEADS, h = bh % N_HEADS;
  #pragma unroll
  for (int dc = 0; dc < 4; ++dc)
    #pragma unroll
    for (int r = 0; r < 8; ++r) {
      const int qg = qRow + rhalf + r;
      const size_t token = (size_t)b * SEQ + qg;
      attnh[token * D_MODEL + h * D_HEAD + dc * 16 + col] = (_Float16)(o[dc][r] / rl[r]);
    }
}

// ---------------- stage 4: output projection ----------------
__global__ __launch_bounds__(256) void proj_gemm(
    const _Float16* __restrict__ ah, const _Float16* __restrict__ WT,
    const float* __restrict__ bias, float* __restrict__ out) {
  __shared__ _Float16 sA[2][128 * 40];
  __shared__ _Float16 sB[2][128 * 40];
  const int m0 = blockIdx.x * 128;
  const int n0 = blockIdx.y * 128;
  const int tid = threadIdx.x;
  const int wave = tid >> 5, lane = tid & 31;
  const int wm = (wave & 3) * 32;
  const int wn = (wave >> 2) * 64;

  v8f acc[2][4] = {};

  const int srow = tid >> 1, sseg = (tid & 1) * 16;
  auto stage = [&](int buf, int k0) {
    const _Float16* ga = ah + (size_t)(m0 + srow) * D_MODEL + k0 + sseg;
    _Float16* la = &sA[buf][srow * 40 + sseg];
    async_copy_b128(la,     ga);
    async_copy_b128(la + 8, ga + 8);
    const _Float16* gb = WT + (size_t)(n0 + srow) * D_MODEL + k0 + sseg;
    _Float16* lb = &sB[buf][srow * 40 + sseg];
    async_copy_b128(lb,     gb);
    async_copy_b128(lb + 8, gb + 8);
  };

  stage(0, 0);
  int cur = 0;
  for (int k0 = 0; k0 < D_MODEL; k0 += 32) {
    wait_async();
    __syncthreads();
    if (k0 + 32 < D_MODEL) stage(cur ^ 1, k0 + 32);
    v16h a0 = load_frag_a(&sA[cur][(wm + 0) * 40], 40);
    v16h a1 = load_frag_a(&sA[cur][(wm + 16) * 40], 40);
    #pragma unroll
    for (int j = 0; j < 4; ++j) {
      v16h b = load_frag_b(&sB[cur][(wn + j * 16) * 40], 40);
      acc[0][j] = WMMA(a0, b, acc[0][j]);
      acc[1][j] = WMMA(a1, b, acc[1][j]);
    }
    cur ^= 1;
  }

  const int col = lane & 15;
  const int rhalf = (lane >> 4) * 8;
  #pragma unroll
  for (int i = 0; i < 2; ++i)
    #pragma unroll
    for (int j = 0; j < 4; ++j)
      #pragma unroll
      for (int r = 0; r < 8; ++r) {
        int m = m0 + wm + i * 16 + rhalf + r;
        int n = n0 + wn + j * 16 + col;
        out[(size_t)m * D_MODEL + n] = acc[i][j][r] + bias[n];
      }
}

// ---------------- launcher ----------------
extern "C" void kernel_launch(void* const* d_in, const int* in_sizes, int n_in,
                              void* d_out, int out_size, void* d_ws, size_t ws_size,
                              hipStream_t stream) {
  const float* x     = (const float*)d_in[0];
  const float* Wqkv  = (const float*)d_in[1];
  const float* bqkv  = (const float*)d_in[2];
  const float* Wproj = (const float*)d_in[3];
  const float* bproj = (const float*)d_in[4];
  float* out = (float*)d_out;

  char* ws = (char*)d_ws;
  size_t off = 0;
  _Float16* xh     = (_Float16*)(ws + off); off += (size_t)NTOK * D_MODEL * 2;      // 6.29 MB
  _Float16* WqkvT  = (_Float16*)(ws + off); off += (size_t)QKV_N * D_MODEL * 2;     // 3.54 MB
  _Float16* WprojT = (_Float16*)(ws + off); off += (size_t)D_MODEL * D_MODEL * 2;   // 1.18 MB
  _Float16* Qh     = (_Float16*)(ws + off); off += (size_t)BATCH * N_HEADS * SEQ * D_HEAD * 2;
  _Float16* Kh     = (_Float16*)(ws + off); off += (size_t)BATCH * N_HEADS * SEQ * D_HEAD * 2;
  _Float16* Vh     = (_Float16*)(ws + off); off += (size_t)BATCH * N_HEADS * SEQ * D_HEAD * 2;
  _Float16* attnh  = (_Float16*)(ws + off); off += (size_t)NTOK * D_MODEL * 2;
  (void)ws_size; (void)in_sizes; (void)n_in; (void)out_size;

  { // conversions
    int nx = NTOK * D_MODEL;
    cvt_f32_f16<<<(nx + 255) / 256, 256, 0, stream>>>(x, xh, nx);
    int nq = D_MODEL * QKV_N;
    cvt_transpose<<<(nq + 255) / 256, 256, 0, stream>>>(Wqkv, WqkvT, D_MODEL, QKV_N);
    int np = D_MODEL * D_MODEL;
    cvt_transpose<<<(np + 255) / 256, 256, 0, stream>>>(Wproj, WprojT, D_MODEL, D_MODEL);
  }

  qkv_gemm<<<dim3(NTOK / 128, QKV_N / 128), 256, 0, stream>>>(xh, WqkvT, bqkv, Qh, Kh, Vh);
  attn_fwd<<<dim3(SEQ / 128, BATCH * N_HEADS), 256, 0, stream>>>(Qh, Kh, Vh, attnh);
  proj_gemm<<<dim3(NTOK / 128, D_MODEL / 128), 256, 0, stream>>>(attnh, WprojT, bproj, out);
}
